// GATBridge_28913719837512
// MI455X (gfx1250) — matrix operand order repaired and checked by reading, not compiled, not run on hardware
//
#include <hip/hip_runtime.h>

// ---------------------------------------------------------------------------
// GAT (2 layers, dense adjacency) for MI455X / gfx1250.
// B=8, N=1024, Din=512, H=4, F=256, HF=1024, Dout=512.
// All GEMMs use v_wmma_f32_16x16x32_f16 (wave32), f16 operands / f32 accum.
// Attention softmax/logits are fused VALU passes; attn stored as f16
// transposed [b,h,d,s] so the aggregation GEMM reads it as a row-major A.
// ---------------------------------------------------------------------------

typedef __attribute__((ext_vector_type(16))) _Float16 v16h;
typedef __attribute__((ext_vector_type(8)))  _Float16 v8h;
typedef __attribute__((ext_vector_type(8)))  float    v8f;

// ---------------------------------------------------------------------------
// Generic NT GEMM: C[M,N] (f32) = A[M,K] (f16, row-major, lda) *
//                                 BT[N,K] (f16, row-major, ldb)^T
// Block = 256 threads = 8 waves (2 M x 4 N), wave tile 64x32 (4x2 frags).
// Per K-step (32): 4 A frags + 2 B frags, 8 WMMAs.
// Batched via blockIdx.z with flexible output offset (b,h) decomposition.
// Requires: M%128==0, N%128==0, K%32==0 (true for all uses here).
//
// WMMA wave32 fragment layouts (CDNA5 ISA 7.12.2):
//  A 16x32 f16 : lane L holds row M=L%16; elems 0..7 = K kk+(L/16)*8+0..7,
//                elems 8..15 = K kk+16+(L/16)*8+0..7  -> two 16B loads.
//  B 32x16 f16 : lane L holds col N=L%16; same K mapping as A
//                -> two 16B loads of BT row.
//  C/D 16x16 f32: lane L col N=L%16; VGPR r row M = r + (L/16)*8.
// ---------------------------------------------------------------------------
__global__ __launch_bounds__(256)
void gemm_nt_f16(const _Float16* __restrict__ A, const _Float16* __restrict__ BT,
                 float* __restrict__ C,
                 int K, int lda, int ldb, int ldc,
                 long aBatch, long bBatch, long cBatchB, long cBatchH, int Hdiv)
{
    const int z  = blockIdx.z;
    const int bb = z / Hdiv;
    const int hh = z % Hdiv;
    A  += (size_t)z  * aBatch;
    BT += (size_t)z  * bBatch;
    C  += (size_t)bb * cBatchB + (size_t)hh * cBatchH;

    const int wid  = threadIdx.x >> 5;
    const int lane = threadIdx.x & 31;
    const int l16  = lane & 15;
    const int half = lane >> 4;

    const int wm  = wid >> 2;                    // 0..1
    const int wn  = wid & 3;                     // 0..3
    const int cm0 = blockIdx.y * 128 + wm * 64;  // wave M origin
    const int cn0 = blockIdx.x * 128 + wn * 32;  // wave N origin

    const v8f zero = {0.f, 0.f, 0.f, 0.f, 0.f, 0.f, 0.f, 0.f};
    v8f acc[4][2];
#pragma unroll
    for (int i = 0; i < 4; ++i)
#pragma unroll
        for (int j = 0; j < 2; ++j) acc[i][j] = zero;

    for (int kk = 0; kk < K; kk += 32) {
        v16h af[4];
#pragma unroll
        for (int i = 0; i < 4; ++i) {
            const _Float16* p = A + (size_t)(cm0 + 16 * i + l16) * lda + kk + half * 8;
            v8h lo = *(const v8h*)p;
            v8h hi = *(const v8h*)(p + 16);
            af[i] = __builtin_shufflevector(lo, hi, 0, 1, 2, 3, 4, 5, 6, 7,
                                            8, 9, 10, 11, 12, 13, 14, 15);
        }
        v16h bf[2];
#pragma unroll
        for (int j = 0; j < 2; ++j) {
            const _Float16* p = BT + (size_t)(cn0 + 16 * j + l16) * ldb + kk + half * 8;
            v8h lo = *(const v8h*)p;
            v8h hi = *(const v8h*)(p + 16);
            bf[j] = __builtin_shufflevector(lo, hi, 0, 1, 2, 3, 4, 5, 6, 7,
                                            8, 9, 10, 11, 12, 13, 14, 15);
        }
#pragma unroll
        for (int i = 0; i < 4; ++i)
#pragma unroll
            for (int j = 0; j < 2; ++j)
                acc[i][j] = __builtin_amdgcn_wmma_f32_16x16x32_f16(
                    false, af[i], false, bf[j], (short)0, acc[i][j], false, false);
    }

#pragma unroll
    for (int i = 0; i < 4; ++i)
#pragma unroll
        for (int j = 0; j < 2; ++j) {
            const int n = cn0 + 16 * j + l16;
#pragma unroll
            for (int r = 0; r < 8; ++r) {
                const int m = cm0 + 16 * i + half * 8 + r;
                C[(size_t)m * ldc + n] = acc[i][j][r];
            }
        }
}

// --------------------------- helper kernels --------------------------------

__global__ void k_f32_to_f16(const float* __restrict__ in, _Float16* __restrict__ out, long n)
{
    long i = (long)blockIdx.x * blockDim.x + threadIdx.x;
    const long stride = (long)gridDim.x * blockDim.x;
    for (; i < n; i += stride) out[i] = (_Float16)in[i];
}

// dst[c][r] (f16) = src[r][c] (f32); rows,cols multiples of 32
__global__ void k_transpose_w(const float* __restrict__ src, _Float16* __restrict__ dst,
                              int rows, int cols)
{
    __shared__ _Float16 tile[32][33];
    const int c0 = blockIdx.x * 32, r0 = blockIdx.y * 32;
    const int tx = threadIdx.x, ty = threadIdx.y;
#pragma unroll
    for (int k = 0; k < 4; ++k)
        tile[ty + 8 * k][tx] = (_Float16)src[(size_t)(r0 + ty + 8 * k) * cols + c0 + tx];
    __syncthreads();
#pragma unroll
    for (int k = 0; k < 4; ++k)
        dst[(size_t)(c0 + ty + 8 * k) * rows + r0 + tx] = tile[tx][ty + 8 * k];
}

// batched f16 transpose: per z=(b*Hdiv+h): dst[z][c][r] = src[b,h][r][c]
__global__ void k_transpose_h(const _Float16* __restrict__ src, _Float16* __restrict__ dst,
                              int srs, int drs, long sB, long sH, long dZ, int Hdiv)
{
    const int z = blockIdx.z;
    const int b = z / Hdiv, h = z % Hdiv;
    src += (size_t)b * sB + (size_t)h * sH;
    dst += (size_t)z * dZ;
    __shared__ _Float16 tile[32][33];
    const int c0 = blockIdx.x * 32, r0 = blockIdx.y * 32;
    const int tx = threadIdx.x, ty = threadIdx.y;
#pragma unroll
    for (int k = 0; k < 4; ++k)
        tile[ty + 8 * k][tx] = src[(size_t)(r0 + ty + 8 * k) * srs + c0 + tx];
    __syncthreads();
#pragma unroll
    for (int k = 0; k < 4; ++k)
        dst[(size_t)(c0 + ty + 8 * k) * drs + r0 + tx] = tile[tx][ty + 8 * k];
}

// maskT[b][d][s] = (adj[b][s][d] > 0) || (s == d)
__global__ void k_build_maskT(const int* __restrict__ adj, unsigned char* __restrict__ maskT, int N)
{
    const int b = blockIdx.z;
    __shared__ unsigned char tile[32][33];
    const int d0 = blockIdx.x * 32, s0 = blockIdx.y * 32;
    const int tx = threadIdx.x, ty = threadIdx.y;
#pragma unroll
    for (int k = 0; k < 4; ++k) {
        const int s = s0 + ty + 8 * k, d = d0 + tx;
        tile[ty + 8 * k][tx] = (unsigned char)((adj[((size_t)b * N + s) * N + d] > 0) || (s == d));
    }
    __syncthreads();
#pragma unroll
    for (int k = 0; k < 4; ++k) {
        const int d = d0 + ty + 8 * k, s = s0 + tx;
        maskT[((size_t)b * N + d) * N + s] = tile[tx][ty + 8 * k];
    }
}

// asrc[b,h,n] = dot(X[b,n,h*F:(h+1)*F], aS[h]); adst likewise.
// grid = B*N blocks, block = H waves of 32.
__global__ void k_rowdot(const float* __restrict__ X, const float* __restrict__ aS,
                         const float* __restrict__ aD, float* __restrict__ oS,
                         float* __restrict__ oD, int N, int H, int F)
{
    const int bn   = blockIdx.x;
    const int h    = threadIdx.x >> 5;
    const int lane = threadIdx.x & 31;
    const float* row = X + (size_t)bn * H * F + (size_t)h * F;
    const float* s   = aS + (size_t)h * F;
    const float* dv  = aD + (size_t)h * F;
    float s1 = 0.f, s2 = 0.f;
    for (int f = lane; f < F; f += 32) {
        const float v = row[f];
        s1 += v * s[f];
        s2 += v * dv[f];
    }
#pragma unroll
    for (int off = 16; off > 0; off >>= 1) {
        s1 += __shfl_xor(s1, off, 32);
        s2 += __shfl_xor(s2, off, 32);
    }
    if (lane == 0) {
        const int b = bn / N, n = bn - b * N;
        const size_t o = ((size_t)b * H + h) * N + n;
        oS[o] = s1;
        oD[o] = s2;
    }
}

// Per (b,h,d): softmax over s of leaky_relu(asrc[s]+adst[d]) with mask.
// N == 1024 (4 x 256 threads). Writes attnT[b,h,d,s] as f16.
__global__ __launch_bounds__(256)
void k_softmax_attn(const float* __restrict__ asrc, const float* __restrict__ adst,
                    const unsigned char* __restrict__ maskT, _Float16* __restrict__ attnT,
                    int N, int H)
{
    const int d   = blockIdx.x;
    const int bh  = blockIdx.y;
    const int b   = bh / H;
    const int tid = threadIdx.x;
    const float* as = asrc + (size_t)bh * N;
    const float  ad = adst[(size_t)bh * N + d];
    const unsigned char* mrow = maskT + ((size_t)b * N + d) * N;

    float ev[4];
    float mx = -3.0e38f;
#pragma unroll
    for (int it = 0; it < 4; ++it) {
        const int s = tid + it * 256;
        float e = as[s] + ad;
        e = e > 0.f ? e : 0.2f * e;           // leaky_relu, slope 0.2
        if (!mrow[s]) e = -1.0e9f;            // NEG_INF
        ev[it] = e;
        mx = fmaxf(mx, e);
    }
    __shared__ float red[256];
    red[tid] = mx;
    __syncthreads();
    for (int off = 128; off > 0; off >>= 1) {
        if (tid < off) red[tid] = fmaxf(red[tid], red[tid + off]);
        __syncthreads();
    }
    mx = red[0];
    __syncthreads();

    float pv[4];
    float sum = 0.f;
#pragma unroll
    for (int it = 0; it < 4; ++it) {
        pv[it] = expf(ev[it] - mx);
        sum += pv[it];
    }
    red[tid] = sum;
    __syncthreads();
    for (int off = 128; off > 0; off >>= 1) {
        if (tid < off) red[tid] += red[tid + off];
        __syncthreads();
    }
    const float inv = 1.0f / red[0];
    _Float16* orow = attnT + ((size_t)bh * N + d) * N;
#pragma unroll
    for (int it = 0; it < 4; ++it) {
        const int s = tid + it * 256;
        orow[s] = (_Float16)(pv[it] * inv);
    }
}

// out_f16[i] = elu(in[i] + bias[i % C])
__global__ void k_bias_elu_f16(const float* __restrict__ in, const float* __restrict__ bias,
                               _Float16* __restrict__ out, long n, int C)
{
    long i = (long)blockIdx.x * blockDim.x + threadIdx.x;
    const long stride = (long)gridDim.x * blockDim.x;
    for (; i < n; i += stride) {
        float v = in[i] + bias[i % C];
        v = v > 0.f ? v : expm1f(v);
        out[i] = (_Float16)v;
    }
}

// out[i] = in[i] + bias[i % C]
__global__ void k_bias_add(const float* __restrict__ in, const float* __restrict__ bias,
                           float* __restrict__ out, long n, int C)
{
    long i = (long)blockIdx.x * blockDim.x + threadIdx.x;
    const long stride = (long)gridDim.x * blockDim.x;
    for (; i < n; i += stride) out[i] = in[i] + bias[i % C];
}

// ---------------------------------------------------------------------------

extern "C" void kernel_launch(void* const* d_in, const int* in_sizes, int n_in,
                              void* d_out, int out_size, void* d_ws, size_t ws_size,
                              hipStream_t stream)
{
    (void)in_sizes; (void)n_in; (void)out_size; (void)ws_size;

    constexpr int B = 8, N = 1024, Din = 512, H = 4, F = 256, HF = 1024, Dout = 512;
    constexpr long BN = (long)B * N;

    const int*   adj = (const int*)  d_in[0];
    const float* x   = (const float*)d_in[1];
    const float* W1  = (const float*)d_in[2];
    const float* as1 = (const float*)d_in[3];
    const float* ad1 = (const float*)d_in[4];
    const float* b1  = (const float*)d_in[5];
    const float* W2  = (const float*)d_in[6];
    const float* as2 = (const float*)d_in[7];
    const float* ad2 = (const float*)d_in[8];
    const float* b2  = (const float*)d_in[9];
    float*       out = (float*)d_out;

    // ---- workspace carve-up ----
    char* w = (char*)d_ws;
    size_t off = 0;
    auto alloc = [&](size_t bytes) -> void* {
        void* p = w + off;
        off += (bytes + 255) & ~(size_t)255;
        return p;
    };
    _Float16* x16    = (_Float16*)alloc(BN * Din * 2);
    _Float16* W1T    = (_Float16*)alloc((size_t)Din * HF * 2);
    _Float16* W2T    = (_Float16*)alloc((size_t)HF * Dout * 2);
    float*    xl     = (float*)   alloc(BN * HF * 4);
    _Float16* xl16   = (_Float16*)alloc(BN * HF * 2);
    _Float16* xlT16  = (_Float16*)alloc(BN * HF * 2);
    float*    asrc1  = (float*)   alloc((size_t)B * H * N * 4);
    float*    adst1  = (float*)   alloc((size_t)B * H * N * 4);
    unsigned char* maskT = (unsigned char*)alloc((size_t)B * N * N);
    _Float16* attnT1 = (_Float16*)alloc((size_t)B * H * N * N * 2);
    float*    out1   = (float*)   alloc(BN * HF * 4);
    _Float16* h1     = (_Float16*)alloc(BN * HF * 2);
    float*    yl     = (float*)   alloc(BN * Dout * 4);
    _Float16* yl16   = (_Float16*)alloc(BN * Dout * 2);
    _Float16* ylT16  = (_Float16*)alloc(BN * Dout * 2);
    float*    asrc2  = (float*)   alloc((size_t)B * N * 4);
    float*    adst2  = (float*)   alloc((size_t)B * N * 4);
    _Float16* attnT2 = (_Float16*)alloc((size_t)B * N * N * 2);
    float*    out2   = (float*)   alloc(BN * Dout * 4);

    const dim3 tb32x8(32, 8);

    // 1) convert x -> f16 ; transpose weights -> f16
    k_f32_to_f16<<<2048, 256, 0, stream>>>(x, x16, BN * Din);
    k_transpose_w<<<dim3(HF / 32, Din / 32), tb32x8, 0, stream>>>(W1, W1T, Din, HF);
    k_transpose_w<<<dim3(Dout / 32, HF / 32), tb32x8, 0, stream>>>(W2, W2T, HF, Dout);

    // 2) xl = x @ W1   [8192 x 1024], K=512
    gemm_nt_f16<<<dim3(HF / 128, BN / 128, 1), 256, 0, stream>>>(
        x16, W1T, xl, Din, Din, Din, HF, 0, 0, 0, 0, 1);

    // 3) xl f16 + per-(b,h) transpose xl^T [f,s]
    k_f32_to_f16<<<2048, 256, 0, stream>>>(xl, xl16, BN * HF);
    k_transpose_h<<<dim3(F / 32, N / 32, B * H), tb32x8, 0, stream>>>(
        xl16, xlT16, HF, N, (long)N * HF, (long)F, (long)F * N, H);

    // 4) attention scalars + mask + softmax (layer 1)
    k_rowdot<<<dim3((unsigned)BN), H * 32, 0, stream>>>(xl, as1, ad1, asrc1, adst1, N, H, F);
    k_build_maskT<<<dim3(N / 32, N / 32, B), tb32x8, 0, stream>>>(adj, maskT, N);
    k_softmax_attn<<<dim3(N, B * H), 256, 0, stream>>>(asrc1, adst1, maskT, attnT1, N, H);

    // 5) out1[b,d,h,f] = attnT1[b,h] @ xl[b,h]   (M=N=1024, N=F=256, K=1024)
    gemm_nt_f16<<<dim3(F / 128, N / 128, B * H), 256, 0, stream>>>(
        attnT1, xlT16, out1, N, N, N, HF,
        (long)N * N, (long)F * N, (long)N * HF, (long)F, H);

    // 6) h1 = elu(out1 + b1) -> f16
    k_bias_elu_f16<<<2048, 256, 0, stream>>>(out1, b1, h1, BN * HF, HF);

    // 7) yl = h1 @ W2   [8192 x 512], K=1024
    gemm_nt_f16<<<dim3(Dout / 128, BN / 128, 1), 256, 0, stream>>>(
        h1, W2T, yl, HF, HF, HF, Dout, 0, 0, 0, 0, 1);

    // 8) layer-2 attention (H=1)
    k_rowdot<<<dim3((unsigned)BN), 32, 0, stream>>>(yl, as2, ad2, asrc2, adst2, N, 1, Dout);
    k_f32_to_f16<<<2048, 256, 0, stream>>>(yl, yl16, BN * Dout);
    k_transpose_h<<<dim3(Dout / 32, N / 32, B), tb32x8, 0, stream>>>(
        yl16, ylT16, Dout, N, (long)N * Dout, 0, (long)Dout * N, 1);
    k_softmax_attn<<<dim3(N, B), 256, 0, stream>>>(asrc2, adst2, maskT, attnT2, N, 1);

    // 9) out2[b,d,f] = attnT2[b] @ yl[b]   (M=1024, N=512, K=1024)
    gemm_nt_f16<<<dim3(Dout / 128, N / 128, B), 256, 0, stream>>>(
        attnT2, ylT16, out2, N, N, N, Dout,
        (long)N * N, (long)Dout * N, (long)N * Dout, 0, 1);

    // 10) final bias
    k_bias_add<<<2048, 256, 0, stream>>>(out2, b2, out, BN * Dout, Dout);
}
